// TreeLSTM_74491912782084
// MI455X (gfx1250) — matrix operand order repaired and checked by reading, not compiled, not run on hardware
//
#include <hip/hip_runtime.h>

typedef __bf16 bf16;
typedef __attribute__((ext_vector_type(16))) __bf16 v16bf;
typedef __attribute__((ext_vector_type(8)))  __bf16 v8bf;
typedef __attribute__((ext_vector_type(8)))  float  v8f;

#define IN_F       128
#define OUT_F      128
#define DEPTH      9
#define N_TREES    128
#define TREE_SIZE  511                 // 2^9 - 1
#define NNODES     (N_TREES * TREE_SIZE)
#define NLAYER     2

#define KT         8                   // 256 / 32 K-tiles
#define IOU_NT     24                  // 384 / 16 N-tiles
#define F_NT       8                   // 128 / 16 N-tiles

// ---------------------------------------------------------------------------
// fp32 -> bf16 (round-to-nearest-even), branch-free.
__device__ __forceinline__ bf16 f2bf(float x) {
    union { float f; unsigned u; } v; v.f = x;
    unsigned r = (v.u + 0x7FFFu + ((v.u >> 16) & 1u)) >> 16;
    unsigned short s = (unsigned short)r;
    return __builtin_bit_cast(bf16, s);
}

__device__ __forceinline__ float sigmoidf(float x) {
    return 1.0f / (1.0f + __expf(-x));
}

// ---------------------------------------------------------------------------
// Pack fp32 weights into bf16 WMMA B-fragment tiles.
//   lane 0..15  -> n = nt*16 + lane,      k = kt*32 + j        (j = 0..15)
//   lane 16..31 -> n = nt*16 + (lane-16), k = kt*32 + 16 + j
// K axis is the fused [x (k<128) | h (k>=128)] input.
// p_iou layout: [L][nt<24][kt<8][512],  p_f layout: [L][nt<8][kt<8][512]
__global__ __launch_bounds__(256) void pack_weights_kernel(
    const float* __restrict__ W_iou, const float* __restrict__ U_iou,
    const float* __restrict__ W_f,   const float* __restrict__ U_f,
    bf16* __restrict__ p_iou, bf16* __restrict__ p_f)
{
    const int iouElems = NLAYER * IOU_NT * KT * 512;   // 196608
    const int fElems   = NLAYER * F_NT   * KT * 512;   //  65536
    int gid = blockIdx.x * 256 + threadIdx.x;
    if (gid < iouElems) {
        int e  = gid;
        int j  = e & 15;
        int l  = (e >> 4) & 31;
        int kt = (e >> 9) & 7;
        int nt = (e >> 12) % IOU_NT;
        int L  = e / (IOU_NT * KT * 512);
        int n  = nt * 16 + (l & 15);
        int k  = kt * 32 + (l >> 4) * 16 + j;
        float v = (k < 128) ? W_iou[(L * 384 + n) * 128 + k]
                            : U_iou[(L * 384 + n) * 128 + (k - 128)];
        p_iou[gid] = f2bf(v);
    } else if (gid < iouElems + fElems) {
        int e  = gid - iouElems;
        int j  = e & 15;
        int l  = (e >> 4) & 31;
        int kt = (e >> 9) & 7;
        int nt = (e >> 12) & 7;
        int L  = e >> 15;
        int n  = nt * 16 + (l & 15);
        int k  = kt * 32 + (l >> 4) * 16 + j;
        float v = (k < 128) ? W_f[(L * 128 + n) * 128 + k]
                            : U_f[(L * 128 + n) * 128 + (k - 128)];
        p_f[e] = f2bf(v);
    }
}

// ---------------------------------------------------------------------------
// Shared per-level body: stage bf16 A panels in LDS, 8 wave32s x 40 N-tiles
// of v_wmma_f32_16x16x32_bf16, then gate math on `count` valid rows.
// sNode/sCl/sCr must be populated (and barrier'd) by the caller.
__device__ __forceinline__ void level_body(
    int tid, int lvl, int layer, int count,
    const float* __restrict__ features,
    const float* __restrict__ b_iou, const float* __restrict__ b_f,
    const bf16* __restrict__ p_iou,  const bf16* __restrict__ p_f,
    float* __restrict__ h, float* __restrict__ c,
    bf16* sAx, bf16* sAhs, bf16* sAhl, bf16* sAhr,
    float* sOut, const int* sNode, const int* sCl, const int* sCr)
{
    // ---- stage A panels (16x128 each) as bf16 into LDS -------------------
    {
        int idx = tid * 8;                  // 8 consecutive floats per thread
        int m   = idx >> 7;
        int col = idx & 127;
        const float* xp = features + (size_t)sNode[m] * 128 + col;
        #pragma unroll
        for (int q = 0; q < 8; ++q) sAx[m * 128 + col + q] = f2bf(xp[q]);
        if (lvl > 0) {
            const float* lp = h + (size_t)sCl[m] * 128 + col;
            const float* rp = h + (size_t)sCr[m] * 128 + col;
            #pragma unroll
            for (int q = 0; q < 8; ++q) {
                float lv = lp[q], rv = rp[q];
                sAhl[m * 128 + col + q] = f2bf(lv);
                sAhr[m * 128 + col + q] = f2bf(rv);
                sAhs[m * 128 + col + q] = f2bf(lv + rv);
            }
        }
    }
    __syncthreads();

    // ---- WMMA GEMMs ------------------------------------------------------
    const int wave = tid >> 5;
    const int lane = tid & 31;
    const int mrow = lane & 15;
    const int half = lane >> 4;             // A-fragment K-half (contiguous k = 16*half + s)

    for (int t = wave; t < 40; t += 8) {
        if (lvl == 0 && t >= 24) break;     // f-gates multiply c_child == 0
        const bf16* Bbase;
        const bf16* Avar;
        if (t < 24)      { Bbase = p_iou + (size_t)(layer * IOU_NT + t)      * KT * 512; Avar = sAhs; }
        else if (t < 32) { Bbase = p_f   + (size_t)(layer * F_NT + (t - 24)) * KT * 512; Avar = sAhl; }
        else             { Bbase = p_f   + (size_t)(layer * F_NT + (t - 32)) * KT * 512; Avar = sAhr; }

        const int ktMax = (lvl == 0) ? 4 : 8;   // h-half of K is zero at leaves
        v8f acc = {};
        for (int kt = 0; kt < ktMax; ++kt) {
            if (kt + 1 < ktMax)                 // WGP-scope, regular-temporal:
                __builtin_prefetch(Bbase + (kt + 1) * 512, 0, 3);

            // A fragment: lane holds M=mrow; element s is k = kloc + s (contiguous order).
            const bf16* abuf = (kt < 4) ? sAx : Avar;
            int kloc = (kt & 3) * 32 + half * 16;
            v8bf a0 = *(const v8bf*)(abuf + mrow * 128 + kloc);
            v8bf a1 = *(const v8bf*)(abuf + mrow * 128 + kloc + 8);
            v16bf afrag = __builtin_shufflevector(
                a0, a1, 0, 1, 2, 3, 4, 5, 6, 7, 8, 9, 10, 11, 12, 13, 14, 15);

            // B fragment: 32 contiguous bytes per lane from packed tile.
            v16bf bfrag = ((const v16bf*)(Bbase + kt * 512))[lane];

            acc = __builtin_amdgcn_wmma_f32_16x16x32_bf16(
                false, afrag, false, bfrag, (short)0, acc, false, false);
        }

        // C/D layout: lane 0..15 -> N=lane, M=r; lane 16..31 -> M=8+r.
        int colbase = t * 16 + (lane & 15);
        int rb      = half * 8;
        #pragma unroll
        for (int r = 0; r < 8; ++r)
            sOut[(rb + r) * 640 + colbase] = acc[r];
    }
    __syncthreads();

    // ---- gate math + state update (valid rows only) ---------------------
    for (int i = tid; i < count * 128; i += 256) {
        int m = i >> 7;
        int j = i & 127;
        float iv = sOut[m * 640 +       j] + b_iou[layer * 384 +       j];
        float ov = sOut[m * 640 + 128 + j] + b_iou[layer * 384 + 128 + j];
        float uv = sOut[m * 640 + 256 + j] + b_iou[layer * 384 + 256 + j];
        float csum = 0.0f;
        if (lvl > 0) {
            float bf  = b_f[layer * 128 + j];
            float flv = sOut[m * 640 + 384 + j] + bf;
            float frv = sOut[m * 640 + 512 + j] + bf;
            float ccl = c[(size_t)sCl[m] * 128 + j];
            float ccr = c[(size_t)sCr[m] * 128 + j];
            csum = sigmoidf(flv) * ccl + sigmoidf(frv) * ccr;
        }
        float cn = sigmoidf(iv) * tanhf(uv) + csum;
        float hn = sigmoidf(ov) * tanhf(cn);
        int node = sNode[m];
        c[(size_t)node * 128 + j] = cn;
        h[(size_t)node * 128 + j] = hn;
    }
}

// ---------------------------------------------------------------------------
// Wide levels (lvl 0..4): one workgroup = 16 active nodes at depth (8-lvl).
__global__ __launch_bounds__(256) void tree_level_kernel(
    const float* __restrict__ features,
    const float* __restrict__ b_iou, const float* __restrict__ b_f,
    const bf16* __restrict__ p_iou,  const bf16* __restrict__ p_f,
    float* __restrict__ h, float* __restrict__ c,
    int lvl, int layer)
{
    __shared__ __align__(16) bf16 sAx [16 * 128];
    __shared__ __align__(16) bf16 sAhs[16 * 128];
    __shared__ __align__(16) bf16 sAhl[16 * 128];
    __shared__ __align__(16) bf16 sAhr[16 * 128];
    __shared__ float sOut[16 * 640];
    __shared__ int   sNode[16], sCl[16], sCr[16];

    const int tid = threadIdx.x;
    const int d   = 8 - lvl;                // heap depth of active nodes

    if (tid < 16) {
        int a    = blockIdx.x * 16 + tid;   // active-node flat index this level
        int tree = a >> d;
        int pos  = a & ((1 << d) - 1);
        int hk   = (1 << d) - 1 + pos;      // heap index within tree
        sNode[tid] = tree * TREE_SIZE + hk;
        int cl = (lvl > 0) ? (tree * TREE_SIZE + 2 * hk + 1) : 0;
        sCl[tid] = cl;
        sCr[tid] = cl + 1;
    }
    __syncthreads();

    level_body(tid, lvl, layer, 16, features, b_iou, b_f, p_iou, p_f, h, c,
               sAx, sAhs, sAhl, sAhr, sOut, sNode, sCl, sCr);
}

// ---------------------------------------------------------------------------
// Serial tail fused: one workgroup per tree runs levels 5..8 (8,4,2,1 nodes)
// with block barriers between levels. 128 WGs/launch instead of 4 launches
// of 64/32/16/8 WGs.
__global__ __launch_bounds__(256) void tree_top_kernel(
    const float* __restrict__ features,
    const float* __restrict__ b_iou, const float* __restrict__ b_f,
    const bf16* __restrict__ p_iou,  const bf16* __restrict__ p_f,
    float* __restrict__ h, float* __restrict__ c,
    int layer)
{
    __shared__ __align__(16) bf16 sAx [16 * 128];
    __shared__ __align__(16) bf16 sAhs[16 * 128];
    __shared__ __align__(16) bf16 sAhl[16 * 128];
    __shared__ __align__(16) bf16 sAhr[16 * 128];
    __shared__ float sOut[16 * 640];
    __shared__ int   sNode[16], sCl[16], sCr[16];

    const int tid  = threadIdx.x;
    const int tree = blockIdx.x;

    for (int lvl = 5; lvl <= 8; ++lvl) {
        __syncthreads();                    // sNode reuse + prev-level h/c WAR
        const int d     = 8 - lvl;
        const int count = 1 << d;           // 8, 4, 2, 1 active nodes
        if (tid < 16) {
            int pos = tid < count ? tid : (count - 1);   // clamp padding rows
            int hk  = (1 << d) - 1 + pos;
            sNode[tid] = tree * TREE_SIZE + hk;
            int cl = tree * TREE_SIZE + 2 * hk + 1;
            sCl[tid] = cl;
            sCr[tid] = cl + 1;
        }
        __syncthreads();

        level_body(tid, lvl, layer, count, features, b_iou, b_f, p_iou, p_f,
                   h, c, sAx, sAhs, sAhl, sAhr, sOut, sNode, sCl, sCr);
    }
}

// ---------------------------------------------------------------------------
// d_out layout: f_h (2,128,128) then f_c (2,128,128). Root = heap index 0.
__global__ __launch_bounds__(256) void root_copy_kernel(
    const float* __restrict__ h, const float* __restrict__ c,
    float* __restrict__ out, int layer)
{
    int gid = blockIdx.x * 256 + threadIdx.x;
    if (gid >= N_TREES * OUT_F) return;
    int t = gid >> 7;
    int j = gid & 127;
    size_t node = (size_t)t * TREE_SIZE;    // root
    out[layer * (N_TREES * OUT_F) + gid]                              = h[node * 128 + j];
    out[NLAYER * (N_TREES * OUT_F) + layer * (N_TREES * OUT_F) + gid] = c[node * 128 + j];
}

// ---------------------------------------------------------------------------
extern "C" void kernel_launch(void* const* d_in, const int* in_sizes, int n_in,
                              void* d_out, int out_size, void* d_ws, size_t ws_size,
                              hipStream_t stream)
{
    const float* features = (const float*)d_in[0];
    const float* W_iou    = (const float*)d_in[1];
    const float* b_iou    = (const float*)d_in[2];
    const float* U_iou    = (const float*)d_in[3];
    const float* W_f      = (const float*)d_in[4];
    const float* b_f      = (const float*)d_in[5];
    const float* U_f      = (const float*)d_in[6];
    // d_in[7..11] (node_order/adjacency/edge_order/treesizes/n_levels) are
    // implied by the fixed complete-heap forest layout; unused.
    float* out = (float*)d_out;

    char* ws = (char*)d_ws;
    bf16*  p_iou = (bf16*)ws;                                  // 393216 B
    bf16*  p_f   = (bf16*)(ws + 393216);                       // 131072 B
    float* h     = (float*)(ws + 524288);                      // N*128 f32
    float* c     = h + (size_t)NNODES * OUT_F;                 // N*128 f32

    // 1) weight repack into WMMA bf16 B-fragments (tiny; L2-resident after)
    pack_weights_kernel<<<(NLAYER * (IOU_NT + F_NT) * KT * 512) / 256, 256, 0, stream>>>(
        W_iou, U_iou, W_f, U_f, p_iou, p_f);

    // 2) two independent TreeLSTM layers: 5 wide levels + fused top, then
    //    root gather. 12 dependent launches total instead of 18.
    for (int L = 0; L < NLAYER; ++L) {
        for (int lvl = 0; lvl < 5; ++lvl) {
            int nblk = 1 << (11 - lvl);     // (128 * 2^(8-lvl)) / 16 nodes/WG
            tree_level_kernel<<<nblk, 256, 0, stream>>>(
                features, b_iou, b_f, p_iou, p_f, h, c, lvl, L);
        }
        tree_top_kernel<<<N_TREES, 256, 0, stream>>>(
            features, b_iou, b_f, p_iou, p_f, h, c, L);
        root_copy_kernel<<<(N_TREES * OUT_F + 255) / 256, 256, 0, stream>>>(h, c, out, L);
    }
}
